// InvertedResidual_19774029431415
// MI455X (gfx1250) — matrix operand order repaired and checked by reading, not compiled, not run on hardware
//
#include <hip/hip_runtime.h>

// ---------------------------------------------------------------------------
// Problem constants
// ---------------------------------------------------------------------------
#define BATCH 16
#define INP   128
#define OUP   128
#define HID   768
#define HH    56
#define WW    56
#define NPIX  (HH * WW)          // 3136
#define TILE  64                 // pixels per workgroup (3136 = 49 * 64)
#define TILES_PER_IMG (NPIX / TILE)   // 49
#define NBLK  (BATCH * TILES_PER_IMG) // 784
#define EPSV  1e-5f

// LDS strides (in bf16 elements), padded to break bank conflicts, 16B aligned
#define XS_STRIDE 136            // 128 K + 8 pad   (272 B per pixel row)
#define HD_STRIDE 776            // 768 K + 8 pad   (1552 B per pixel row)

typedef __attribute__((ext_vector_type(16))) __bf16 v16bf;
typedef __attribute__((ext_vector_type(8)))  float  v8f;
typedef __attribute__((ext_vector_type(4)))  unsigned int u32x4;
typedef __attribute__((ext_vector_type(4)))  float  f32x4;
typedef __attribute__((ext_vector_type(8)))  unsigned short u16x8;

union FragBF {                   // 16x32 / 32x16 bf16 fragment = 32 bytes
    v16bf v;
    u32x4 q[2];
};

__device__ __forceinline__ float bf2f(unsigned short u) {
    unsigned int t = ((unsigned int)u) << 16;
    return __builtin_bit_cast(float, t);
}

// ---------------------------------------------------------------------------
// Prep: fold BN params, convert weights fp32 -> bf16
// ---------------------------------------------------------------------------
__global__ void prep_kernel(const float* __restrict__ w1,
                            const float* __restrict__ w2,
                            const float* __restrict__ g1, const float* __restrict__ b1,
                            const float* __restrict__ m1, const float* __restrict__ v1,
                            const float* __restrict__ g2, const float* __restrict__ b2,
                            const float* __restrict__ m2, const float* __restrict__ v2,
                            const float* __restrict__ g3, const float* __restrict__ b3,
                            const float* __restrict__ m3, const float* __restrict__ v3,
                            __bf16* __restrict__ w1b, __bf16* __restrict__ w2b,
                            float* __restrict__ s1, float* __restrict__ t1,
                            float* __restrict__ s2, float* __restrict__ t2,
                            float* __restrict__ s3, float* __restrict__ t3) {
    int idx = blockIdx.x * blockDim.x + threadIdx.x;
    if (idx < HID * INP) {
        w1b[idx] = (__bf16)w1[idx];
        w2b[idx] = (__bf16)w2[idx];   // same element count (128*768)
    }
    if (idx < HID) {
        float s = g1[idx] * __frsqrt_rn(v1[idx] + EPSV);
        s1[idx] = s;  t1[idx] = b1[idx] - m1[idx] * s;
        float q = g2[idx] * __frsqrt_rn(v2[idx] + EPSV);
        s2[idx] = q;  t2[idx] = b2[idx] - m2[idx] * q;
    }
    if (idx < OUP) {
        float s = g3[idx] * __frsqrt_rn(v3[idx] + EPSV);
        s3[idx] = s;  t3[idx] = b3[idx] - m3[idx] * s;
    }
}

// ---------------------------------------------------------------------------
// Stage 1: 1x1 expand GEMM (bf16 WMMA) + BN1 + ReLU6 + dilated mask
// One workgroup = 64 consecutive pixels of one image, 8 waves.
// ---------------------------------------------------------------------------
__global__ void __launch_bounds__(256)
expand_kernel(const float* __restrict__ x,
              const __bf16* __restrict__ w1b,
              const float* __restrict__ s1, const float* __restrict__ t1,
              const int* __restrict__ mask,
              __bf16* __restrict__ h) {
    __shared__ __align__(16) __bf16 xs[TILE * XS_STRIDE]; // [pixel][K] bf16
    __shared__ float md[TILE];

    const int tid  = threadIdx.x;
    const int blk  = blockIdx.x;
    const int b    = blk / TILES_PER_IMG;
    const int rem0 = (blk % TILES_PER_IMG) * TILE;

    // --- dilated mask for this pixel tile ---
    if (tid < TILE) {
        int rem = rem0 + tid;
        int y = rem / WW, xp = rem % WW;
        int mm = 0;
        const int* mp = mask + b * NPIX;
        #pragma unroll
        for (int dy = -1; dy <= 1; ++dy) {
            #pragma unroll
            for (int dx = -1; dx <= 1; ++dx) {
                int ny = y + dy, nx = xp + dx;
                if (ny >= 0 && ny < HH && nx >= 0 && nx < WW) {
                    int mv = mp[ny * WW + nx];
                    mm = mv > mm ? mv : mm;
                }
            }
        }
        md[tid] = (float)mm;
    }

    // --- stage x tile into LDS, transposed to K-contiguous bf16 ---
    {
        const float* xb = x + (size_t)b * INP * NPIX + rem0;
        #pragma unroll
        for (int i = 0; i < 8; ++i) {
            int idx = tid + i * 256;        // (c, t4)
            int c  = idx >> 4;
            int t4 = (idx & 15) << 2;
            f32x4 vv = *(const f32x4*)(xb + (size_t)c * NPIX + t4);
            xs[(t4 + 0) * XS_STRIDE + c] = (__bf16)vv.x;
            xs[(t4 + 1) * XS_STRIDE + c] = (__bf16)vv.y;
            xs[(t4 + 2) * XS_STRIDE + c] = (__bf16)vv.z;
            xs[(t4 + 3) * XS_STRIDE + c] = (__bf16)vv.w;
        }
    }
    __syncthreads();

    const int wave = tid >> 5;
    const int lane = tid & 31;
    const int lrow = lane & 15;
    const int lhi  = lane >> 4;       // 0 or 1
    __bf16* hb = h + (size_t)b * HID * NPIX + rem0;

    // each wave: 6 row-blocks of 16 hidden channels x 64 pixels
    for (int rb = wave; rb < HID / 16; rb += 8) {
        const int m0 = rb * 16;
        v8f acc[4] = {};
        #pragma unroll
        for (int kb = 0; kb < INP / 32; ++kb) {
            FragBF a;
            const u32x4* ap = (const u32x4*)(w1b + (size_t)(m0 + lrow) * INP
                                             + kb * 32 + lhi * 16);
            a.q[0] = ap[0];           // K = kbase .. kbase+7
            a.q[1] = ap[1];           // K = kbase+8 .. kbase+15
            __builtin_prefetch(ap + 4, 0, 3);   // next kb A-tile
            #pragma unroll
            for (int n = 0; n < 4; ++n) {
                FragBF bf;
                const u32x4* bp = (const u32x4*)(xs + (n * 16 + lrow) * XS_STRIDE
                                                 + kb * 32 + lhi * 16);
                bf.q[0] = bp[0];
                bf.q[1] = bp[1];
                acc[n] = __builtin_amdgcn_wmma_f32_16x16x32_bf16(
                    false, a.v, false, bf.v, (short)0, acc[n], false, false);
            }
        }
        // epilogue: BN1 + ReLU6 + dilated-mask, store bf16
        float sr[8], tr[8];
        int rows[8];
        #pragma unroll
        for (int j = 0; j < 8; ++j) {
            int r = m0 + j + lhi * 8;
            rows[j] = r; sr[j] = s1[r]; tr[j] = t1[r];
        }
        #pragma unroll
        for (int n = 0; n < 4; ++n) {
            int col = n * 16 + lrow;
            float mdv = md[col];
            #pragma unroll
            for (int j = 0; j < 8; ++j) {
                float v = acc[n][j] * sr[j] + tr[j];
                v = fminf(fmaxf(v, 0.0f), 6.0f) * mdv;
                hb[(size_t)rows[j] * NPIX + col] = (__bf16)v;
            }
        }
    }
}

// ---------------------------------------------------------------------------
// Stage 2: depthwise 3x3 + BN2 + ReLU6 + mask  ->  1x1 project GEMM (bf16
// WMMA) + BN3 + mask + residual add.
//
// Depthwise phase is vectorized: each thread produces 8 contiguous pixels of
// one channel. 56 % 8 == 0, so an 8-pixel group never straddles an image row;
// each conv row needs one aligned b128 (8 bf16) + two edge u16 loads.
// ---------------------------------------------------------------------------
__global__ void __launch_bounds__(256)
dwproj_kernel(const __bf16* __restrict__ h,
              const float* __restrict__ wdw,
              const float* __restrict__ s2, const float* __restrict__ t2,
              const __bf16* __restrict__ w2b,
              const float* __restrict__ s3, const float* __restrict__ t3,
              const int* __restrict__ mask,
              const float* __restrict__ x,
              float* __restrict__ out) {
    __shared__ __align__(16) __bf16 hd[TILE * HD_STRIDE]; // [pixel][K] bf16
    __shared__ float mfs[TILE];

    const int tid  = threadIdx.x;
    const int blk  = blockIdx.x;
    const int b    = blk / TILES_PER_IMG;
    const int rem0 = (blk % TILES_PER_IMG) * TILE;

    if (tid < TILE)
        mfs[tid] = (float)mask[(size_t)b * NPIX + rem0 + tid];
    __syncthreads();

    // --- depthwise 3x3 into LDS (K-contiguous per pixel) ---
    // 6144 tasks (8 pixel-groups x 768 channels) over 256 threads -> 24 each
    {
        const unsigned short* hbase =
            (const unsigned short*)(h + (size_t)b * HID * NPIX);
        #pragma unroll 4
        for (int it = 0; it < 24; ++it) {
            const int task = tid + it * 256;
            const int g  = task & 7;          // pixel group within tile
            const int c  = task >> 3;         // hidden channel
            const int p0 = rem0 + g * 8;      // first pixel (multiple of 8)
            const int y  = p0 / WW;
            const int x0 = p0 % WW;           // in {0,8,...,48}
            const unsigned short* hp = hbase + (size_t)c * NPIX;

            const bool hasL = (x0 > 0), hasR = (x0 < WW - 8);
            const bool hasT = (y > 0),  hasB = (y < HH - 1);

            // window rows: r?[i] corresponds to pixel (p0 - 1 + i) in that row
            float r0[10], r1[10], r2[10];
            // center row
            {
                u16x8 cm = *(const u16x8*)(hp + p0);
                r1[0] = hasL ? bf2f(hp[p0 - 1]) : 0.0f;
                #pragma unroll
                for (int j = 0; j < 8; ++j) r1[j + 1] = bf2f(cm[j]);
                r1[9] = hasR ? bf2f(hp[p0 + 8]) : 0.0f;
            }
            // top row
            if (hasT) {
                u16x8 tm = *(const u16x8*)(hp + p0 - WW);
                r0[0] = hasL ? bf2f(hp[p0 - WW - 1]) : 0.0f;
                #pragma unroll
                for (int j = 0; j < 8; ++j) r0[j + 1] = bf2f(tm[j]);
                r0[9] = hasR ? bf2f(hp[p0 - WW + 8]) : 0.0f;
            } else {
                #pragma unroll
                for (int j = 0; j < 10; ++j) r0[j] = 0.0f;
            }
            // bottom row
            if (hasB) {
                u16x8 bm = *(const u16x8*)(hp + p0 + WW);
                r2[0] = hasL ? bf2f(hp[p0 + WW - 1]) : 0.0f;
                #pragma unroll
                for (int j = 0; j < 8; ++j) r2[j + 1] = bf2f(bm[j]);
                r2[9] = hasR ? bf2f(hp[p0 + WW + 8]) : 0.0f;
            } else {
                #pragma unroll
                for (int j = 0; j < 10; ++j) r2[j] = 0.0f;
            }

            const float* wp = wdw + c * 9;
            float w0 = wp[0], w1 = wp[1], w2v = wp[2];
            float w3 = wp[3], w4 = wp[4], w5 = wp[5];
            float w6 = wp[6], w7 = wp[7], w8 = wp[8];
            const float sc = s2[c], tc = t2[c];

            #pragma unroll
            for (int j = 0; j < 8; ++j) {
                float acc = r0[j] * w0 + r0[j + 1] * w1 + r0[j + 2] * w2v
                          + r1[j] * w3 + r1[j + 1] * w4 + r1[j + 2] * w5
                          + r2[j] * w6 + r2[j + 1] * w7 + r2[j + 2] * w8;
                float v = acc * sc + tc;
                v = fminf(fmaxf(v, 0.0f), 6.0f) * mfs[g * 8 + j];
                hd[(g * 8 + j) * HD_STRIDE + c] = (__bf16)v;
            }
        }
    }
    __syncthreads();

    // --- project GEMM: out[128, 64] = w2[128,768] x hd[768,64] ---
    const int wave = tid >> 5;
    const int lane = tid & 31;
    const int lrow = lane & 15;
    const int lhi  = lane >> 4;
    const int m0   = wave * 16;       // 8 waves cover 128 output channels

    v8f acc[4] = {};
    #pragma unroll
    for (int kb = 0; kb < HID / 32; ++kb) {
        FragBF a;
        const u32x4* ap = (const u32x4*)(w2b + (size_t)(m0 + lrow) * HID
                                         + kb * 32 + lhi * 16);
        a.q[0] = ap[0];
        a.q[1] = ap[1];
        __builtin_prefetch(ap + 4, 0, 3);       // next kb A-tile
        #pragma unroll
        for (int n = 0; n < 4; ++n) {
            FragBF bf;
            const u32x4* bp = (const u32x4*)(hd + (n * 16 + lrow) * HD_STRIDE
                                             + kb * 32 + lhi * 16);
            bf.q[0] = bp[0];
            bf.q[1] = bp[1];
            acc[n] = __builtin_amdgcn_wmma_f32_16x16x32_bf16(
                false, a.v, false, bf.v, (short)0, acc[n], false, false);
        }
    }

    // epilogue: BN3 + mask + residual
    float sr[8], tr[8];
    int rows[8];
    #pragma unroll
    for (int j = 0; j < 8; ++j) {
        int r = m0 + j + lhi * 8;
        rows[j] = r; sr[j] = s3[r]; tr[j] = t3[r];
    }
    const float* xb = x   + (size_t)b * OUP * NPIX + rem0;
    float*       ob = out + (size_t)b * OUP * NPIX + rem0;
    #pragma unroll
    for (int n = 0; n < 4; ++n) {
        int col = n * 16 + lrow;
        float mfv = mfs[col];
        #pragma unroll
        for (int j = 0; j < 8; ++j) {
            size_t idx = (size_t)rows[j] * NPIX + col;
            float v = (acc[n][j] * sr[j] + tr[j]) * mfv;
            ob[idx] = xb[idx] + v;
        }
    }
}

// ---------------------------------------------------------------------------
// Launcher
// ---------------------------------------------------------------------------
extern "C" void kernel_launch(void* const* d_in, const int* in_sizes, int n_in,
                              void* d_out, int out_size, void* d_ws, size_t ws_size,
                              hipStream_t stream) {
    (void)in_sizes; (void)n_in; (void)out_size; (void)ws_size;

    const float* x    = (const float*)d_in[0];
    const float* w1   = (const float*)d_in[1];
    const float* g1   = (const float*)d_in[2];
    const float* b1   = (const float*)d_in[3];
    const float* m1   = (const float*)d_in[4];
    const float* v1   = (const float*)d_in[5];
    const float* wdw  = (const float*)d_in[6];
    const float* g2   = (const float*)d_in[7];
    const float* b2   = (const float*)d_in[8];
    const float* m2   = (const float*)d_in[9];
    const float* v2   = (const float*)d_in[10];
    const float* w2   = (const float*)d_in[11];
    const float* g3   = (const float*)d_in[12];
    const float* b3   = (const float*)d_in[13];
    const float* m3   = (const float*)d_in[14];
    const float* v3   = (const float*)d_in[15];
    const int*   mask = (const int*)d_in[16];
    float*       out  = (float*)d_out;

    // workspace carve-up (bytes)
    char* ws = (char*)d_ws;
    __bf16* w1b = (__bf16*)(ws + 0);                       // 768*128*2
    __bf16* w2b = (__bf16*)(ws + 196608);                  // 128*768*2
    float*  s1  = (float*)(ws + 393216);                   // 768*4
    float*  t1  = (float*)(ws + 396288);
    float*  s2  = (float*)(ws + 399360);
    float*  t2  = (float*)(ws + 402432);
    float*  s3  = (float*)(ws + 405504);                   // 128*4
    float*  t3  = (float*)(ws + 406016);
    __bf16* h   = (__bf16*)(ws + 406528);                  // 16*768*3136*2 (~73.5 MB, L2-resident)

    prep_kernel<<<(HID * INP + 255) / 256, 256, 0, stream>>>(
        w1, w2, g1, b1, m1, v1, g2, b2, m2, v2, g3, b3, m3, v3,
        w1b, w2b, s1, t1, s2, t2, s3, t3);

    expand_kernel<<<NBLK, 256, 0, stream>>>(x, w1b, s1, t1, mask, h);

    dwproj_kernel<<<NBLK, 256, 0, stream>>>(h, wdw, s2, t2, w2b, s3, t3,
                                            mask, x, out);
}
